// GATSelfAttention_40999757807671
// MI455X (gfx1250) — matrix operand order repaired and checked by reading, not compiled, not run on hardware
//
#include <hip/hip_runtime.h>

// ---------------------------------------------------------------------------
// GAT self-attention forward for gfx1250 (MI455X), f16 WMMA w/ f32 accumulate.
// L=128, B=2, E=768, H=12, Dh=64, BH=24, ROWS=L*B=256.
// ---------------------------------------------------------------------------

typedef _Float16 f16_t;
typedef __attribute__((ext_vector_type(16))) _Float16 v16h;
typedef __attribute__((ext_vector_type(2)))  _Float16 h2v;
typedef __attribute__((ext_vector_type(8)))  float    v8f;

#define E_    768
#define EH_   384
#define L_    128
#define B_    2
#define H_    12
#define DH_   64
#define BH_   24
#define ROWS_ 256

// ----- WMMA fragment loaders (ISA 7.12.2, 16-bit A 16x32 / B 32x16) --------
// A: lane&15 = M row; lane>>4 selects K-halves {0..7,16..23} vs {8..15,24..31}
__device__ __forceinline__ v16h load_fragA(const f16_t* row, int k0, int h16) {
  v16h f;
#pragma unroll
  for (int v = 0; v < 8; ++v) {
    int koff = k0 + ((v & 3) << 1) + ((v >> 2) << 4) + (h16 << 3);
    h2v p = *(const h2v*)(row + koff);
    f[2 * v] = p[0];
    f[2 * v + 1] = p[1];
  }
  return f;
}
// B: lane&15 = N column; lane>>4 selects K-halves {0..7,16..23} vs {8..15,24..31}
__device__ __forceinline__ v16h load_fragB(const f16_t* col, int k0, int h16) {
  v16h f;
#pragma unroll
  for (int v = 0; v < 8; ++v) {
    int koff = k0 + ((v & 3) << 1) + ((v >> 2) << 4) + (h16 << 3);
    h2v p = *(const h2v*)(col + koff);
    f[2 * v] = p[0];
    f[2 * v + 1] = p[1];
  }
  return f;
}

// ----- Generic wave32 WMMA GEMM:  C = A*B1^T (+ Aprev*B2^T) + bias ---------
// A rows are K-contiguous (lda), B rows are weight rows W[n, :] (ldb).
// One wave per 16x16 C tile; grid = (N/16, M/16, batch).
// cMode 0: C base = bz*bsC. cMode 1: attention scatter base
//          = (bz/12)*768 + (bz%12)*64 (a[l, b, h*64+d] layout, ldc=1536).
__global__ __launch_bounds__(32)
void k_gemm_wmma(const f16_t* __restrict__ A, long long bsA, int lda,
                 const f16_t* __restrict__ B1, const f16_t* __restrict__ B2,
                 long long bsB, int ldb,
                 const float* __restrict__ bias,
                 float* __restrict__ Cf, f16_t* __restrict__ Ch,
                 int ldc, long long bsC, int cMode,
                 int K, int prevDelta) {
  const int lane = threadIdx.x & 31;
  const int h16  = lane >> 4;
  const int ln   = lane & 15;
  const int bm   = blockIdx.y << 4;
  const int bn   = blockIdx.x << 4;
  const int bz   = blockIdx.z;

  const f16_t* Ab   = A + (long long)bz * bsA;
  const f16_t* Arow = Ab + (long long)(bm + ln) * lda;
  const f16_t* Brow = B1 + (long long)bz * bsB + (long long)(bn + ln) * ldb;

  v8f acc = {0.f, 0.f, 0.f, 0.f, 0.f, 0.f, 0.f, 0.f};

  for (int k0 = 0; k0 < K; k0 += 32) {
    __builtin_prefetch(Arow + k0 + 64, 0, 0);   // global_prefetch_b8
    v16h af = load_fragA(Arow, k0, h16);
    v16h bf = load_fragB(Brow, k0, h16);
    acc = __builtin_amdgcn_wmma_f32_16x16x32_f16(
        false, af, false, bf, (short)0, acc, false, false);
  }

  if (B2 != nullptr) {  // second tap: A rows shifted back by prevDelta (conv k=2)
    const int pm = bm + ln - prevDelta;
    const _Float16 sc = (pm >= 0) ? (_Float16)1.0f : (_Float16)0.0f;
    const f16_t* Ap = Ab + (long long)(pm >= 0 ? pm : 0) * lda;
    const f16_t* B2row = B2 + (long long)bz * bsB + (long long)(bn + ln) * ldb;
    for (int k0 = 0; k0 < K; k0 += 32) {
      v16h af = load_fragA(Ap, k0, h16);
      af = af * sc;                         // zero pad for t-1 < 0 rows
      v16h bf = load_fragB(B2row, k0, h16);
      acc = __builtin_amdgcn_wmma_f32_16x16x32_f16(
          false, af, false, bf, (short)0, acc, false, false);
    }
  }

  const long long cbase =
      (cMode == 0) ? (long long)bz * bsC
                   : (long long)((bz / 12) * 768 + (bz % 12) * 64);
  const int col = bn + ln;
  const float bv = bias ? bias[col] : 0.0f;
#pragma unroll
  for (int c = 0; c < 8; ++c) {   // C layout: VGPR c -> M = c + 8*h16, N = ln
    int row = bm + c + (h16 << 3);
    float v = acc[c] + bv;
    long long idx = cbase + (long long)row * ldc + col;
    if (Cf) Cf[idx] = v;
    if (Ch) Ch[idx] = (f16_t)v;
  }
}

// ----- elementwise converters ----------------------------------------------
__global__ void k_cvt(f16_t* __restrict__ dst, const float* __restrict__ src, int n) {
  int i = blockIdx.x * 256 + threadIdx.x;
  if (i < n) dst[i] = (f16_t)src[i];
}
// conv_w (O,I,2) -> two (O,I) f16 matrices (taps 0,1)
__global__ void k_cvt_pair(f16_t* __restrict__ d0, f16_t* __restrict__ d1,
                           const float* __restrict__ src, int n) {
  int i = blockIdx.x * 256 + threadIdx.x;
  if (i < n) {
    d0[i] = (f16_t)src[2 * i];
    d1[i] = (f16_t)src[2 * i + 1];
  }
}
// deconv_w (I,O,2) -> two transposed (O,I) f16 matrices
__global__ void k_cvt_deconvT(f16_t* __restrict__ d0, f16_t* __restrict__ d1,
                              const float* __restrict__ src) {
  int idx = blockIdx.x * 256 + threadIdx.x;
  if (idx < E_ * E_) {
    int o = idx / E_, i = idx % E_;
    const float* s = src + (size_t)(i * E_ + o) * 2;
    d0[idx] = (f16_t)s[0];
    d1[idx] = (f16_t)s[1];
  }
}

// ----- factorized score tables ---------------------------------------------
// Aq[m,x]=dot(qe[m,x,:],w[:64]); Bk[m,l]=dot(ke[m,l,:],w[64:128]);
// Ce[i,c]=dot(entity[i,b,h*64:],w[128:192]) with c=b*12+h.
__global__ void k_rowdots(const float* __restrict__ q_proj,
                          const float* __restrict__ k_proj,
                          const float* __restrict__ e_proj,
                          const float* __restrict__ entity,
                          const float* __restrict__ attn_w,
                          float* __restrict__ Aq, float* __restrict__ Bk,
                          float* __restrict__ Ce) {
  int t = blockIdx.x * 256 + threadIdx.x;
  if (t >= 3 * BH_ * L_) return;
  int which = t / (BH_ * L_);
  int r = t % (BH_ * L_);
  float s = 0.f;
  if (which < 2) {                     // Aq / Bk : r = m*128 + x
    int m = r / L_, x = r % L_;
    int b = m / H_, j = m % H_;
    const float* w = attn_w + which * DH_;
    const float* src = (j < 6)
        ? ((which == 0 ? q_proj : k_proj) + (size_t)(x * B_ + b) * EH_ + j * DH_)
        : (e_proj + (size_t)(x * B_ + b) * EH_ + (j - 6) * DH_);
#pragma unroll 8
    for (int d = 0; d < DH_; ++d) s += src[d] * w[d];
    (which == 0 ? Aq : Bk)[r] = s;
  } else {                             // Ce : r = i*24 + c
    int i = r / BH_, c = r % BH_;
    int b = c / H_, h = c % H_;
    const float* src = entity + (size_t)(i * B_ + b) * E_ + h * DH_;
    const float* w = attn_w + 2 * DH_;
#pragma unroll 8
    for (int d = 0; d < DH_; ++d) s += src[d] * w[d];
    Ce[r] = s;
  }
}

// ----- pack V/E into veT[bh][d][l] (f16, N x K row-major for WMMA B) -------
__global__ void k_pack_ve(const float* __restrict__ v_proj,
                          const float* __restrict__ e_proj,
                          f16_t* __restrict__ veT) {
  int t = blockIdx.x * 256 + threadIdx.x;
  if (t >= BH_ * DH_ * L_) return;
  int bh = t / (DH_ * L_);
  int rem = t % (DH_ * L_);
  int d = rem / L_, l = rem % L_;
  int b = bh / H_, j = bh % H_;
  float v = (j < 6) ? v_proj[(size_t)(l * B_ + b) * EH_ + j * DH_ + d]
                    : e_proj[(size_t)(l * B_ + b) * EH_ + (j - 6) * DH_ + d];
  veT[t] = (f16_t)v;
}

// ----- fused score assembly + leaky-relu + softmax (one wave per row) ------
__global__ __launch_bounds__(32)
void k_scores_softmax(const float* __restrict__ Aq, const float* __restrict__ Bk,
                      const float* __restrict__ Ce, const float* __restrict__ attn_b,
                      f16_t* __restrict__ probs) {
  const int row = blockIdx.x;            // row = bh*128 + x, 3072 rows
  const int lane = threadIdx.x & 31;
  const int gBase = row * L_;
  const int mK = row % BH_;              // ke index (g/128)%24 is row-constant
  const float ab = attn_b[0];
  float s[4];
  float mx = -3.0e38f;
#pragma unroll
  for (int u = 0; u < 4; ++u) {
    int y = lane + u * 32;
    int g = gBase + y;
    int m   = g % BH_;                   // qe lane / edge column
    int i   = g / 3072;                  // qe x2 == edge i
    int j   = (g % 3072) / BH_;          // edge j
    float sc = Aq[m * L_ + i] + Bk[mK * L_ + y] +
               Ce[i * BH_ + m] - Ce[j * BH_ + m] + ab;
    sc = (sc >= 0.f) ? sc : 0.01f * sc;  // leaky_relu
    s[u] = sc;
    mx = fmaxf(mx, sc);
  }
#pragma unroll
  for (int off = 16; off >= 1; off >>= 1) mx = fmaxf(mx, __shfl_xor(mx, off, 32));
  float sum = 0.f;
#pragma unroll
  for (int u = 0; u < 4; ++u) { s[u] = __expf(s[u] - mx); sum += s[u]; }
#pragma unroll
  for (int off = 16; off >= 1; off >>= 1) sum += __shfl_xor(sum, off, 32);
  float inv = 1.0f / sum;
#pragma unroll
  for (int u = 0; u < 4; ++u)
    probs[gBase + lane + u * 32] = (f16_t)(s[u] * inv);
}

// ---------------------------------------------------------------------------
extern "C" void kernel_launch(void* const* d_in, const int* in_sizes, int n_in,
                              void* d_out, int out_size, void* d_ws, size_t ws_size,
                              hipStream_t stream) {
  const float* query    = (const float*)d_in[0];
  const float* entity   = (const float*)d_in[1];
  const float* conv_w   = (const float*)d_in[2];
  const float* conv_b   = (const float*)d_in[3];
  const float* q_w      = (const float*)d_in[4];
  const float* q_b      = (const float*)d_in[5];
  const float* k_w      = (const float*)d_in[6];
  const float* k_b      = (const float*)d_in[7];
  const float* v_w      = (const float*)d_in[8];
  const float* v_b      = (const float*)d_in[9];
  const float* e_w      = (const float*)d_in[10];
  const float* e_b      = (const float*)d_in[11];
  const float* attn_w   = (const float*)d_in[12];
  const float* attn_b   = (const float*)d_in[13];
  const float* deconv_w = (const float*)d_in[14];
  const float* deconv_b = (const float*)d_in[15];
  const float* out_w    = (const float*)d_in[16];
  const float* out_b    = (const float*)d_in[17];
  float* out = (float*)d_out;

  // ---- workspace carve (≈12 MB) ----
  char* base = (char*)d_ws;
  size_t off = 0;
  auto alloc = [&](size_t bytes) -> void* {
    off = (off + 255) & ~(size_t)255;
    void* r = base + off;
    off += bytes;
    return r;
  };
  const int NLBE = L_ * B_ * E_;     // 196608
  const int NW   = E_ * E_;          // 589824
  const int NWH  = EH_ * E_;         // 294912
  f16_t* q16    = (f16_t*)alloc(NLBE * 2);
  f16_t* ent16  = (f16_t*)alloc(NLBE * 2);
  f16_t* w0_16  = (f16_t*)alloc(NW * 2);
  f16_t* w1_16  = (f16_t*)alloc(NW * 2);
  f16_t* qw16   = (f16_t*)alloc(NWH * 2);
  f16_t* kw16   = (f16_t*)alloc(NWH * 2);
  f16_t* vw16   = (f16_t*)alloc(NWH * 2);
  f16_t* ew16   = (f16_t*)alloc(NWH * 2);
  f16_t* outw16 = (f16_t*)alloc(NW * 2);
  f16_t* dw0_16 = (f16_t*)alloc(NW * 2);
  f16_t* dw1_16 = (f16_t*)alloc(NW * 2);
  f16_t* nq16   = (f16_t*)alloc(NLBE * 2);
  f16_t* a16    = (f16_t*)alloc(NLBE * 2);
  f16_t* y16    = (f16_t*)alloc(NLBE * 2);
  f16_t* probs16= (f16_t*)alloc(BH_ * L_ * L_ * 2);
  f16_t* veT16  = (f16_t*)alloc(BH_ * DH_ * L_ * 2);
  float* q_proj = (float*)alloc(ROWS_ * EH_ * 4);
  float* k_proj = (float*)alloc(ROWS_ * EH_ * 4);
  float* v_proj = (float*)alloc(ROWS_ * EH_ * 4);
  float* e_proj = (float*)alloc(ROWS_ * EH_ * 4);
  float* Aq     = (float*)alloc(BH_ * L_ * 4);
  float* Bk     = (float*)alloc(BH_ * L_ * 4);
  float* Ce     = (float*)alloc(L_ * BH_ * 4);

  auto cvblk = [](int n) { return dim3((n + 255) / 256); };

  // ---- 0) f16 staging ----
  k_cvt<<<cvblk(NLBE), 256, 0, stream>>>(q16, query, NLBE);
  k_cvt<<<cvblk(NLBE), 256, 0, stream>>>(ent16, entity, NLBE);
  k_cvt<<<cvblk(NWH), 256, 0, stream>>>(qw16, q_w, NWH);
  k_cvt<<<cvblk(NWH), 256, 0, stream>>>(kw16, k_w, NWH);
  k_cvt<<<cvblk(NWH), 256, 0, stream>>>(vw16, v_w, NWH);
  k_cvt<<<cvblk(NWH), 256, 0, stream>>>(ew16, e_w, NWH);
  k_cvt<<<cvblk(NW), 256, 0, stream>>>(outw16, out_w, NW);
  k_cvt_pair<<<cvblk(NW), 256, 0, stream>>>(w0_16, w1_16, conv_w, NW);
  k_cvt_deconvT<<<cvblk(NW), 256, 0, stream>>>(dw0_16, dw1_16, deconv_w);

  // ---- 1) conv1d as 2-tap GEMM: nq = q@W1^T + qprev@W0^T + b -> nq16 ----
  k_gemm_wmma<<<dim3(E_ / 16, ROWS_ / 16, 1), 32, 0, stream>>>(
      q16, 0, E_, w1_16, w0_16, 0, E_, conv_b,
      nullptr, nq16, E_, 0, 0, E_, B_);

  // ---- 2) projections e(entity), q/k/v(nq) -> f32 ----
  k_gemm_wmma<<<dim3(EH_ / 16, ROWS_ / 16, 1), 32, 0, stream>>>(
      ent16, 0, E_, ew16, nullptr, 0, E_, e_b, e_proj, nullptr, EH_, 0, 0, E_, 0);
  k_gemm_wmma<<<dim3(EH_ / 16, ROWS_ / 16, 1), 32, 0, stream>>>(
      nq16, 0, E_, qw16, nullptr, 0, E_, q_b, q_proj, nullptr, EH_, 0, 0, E_, 0);
  k_gemm_wmma<<<dim3(EH_ / 16, ROWS_ / 16, 1), 32, 0, stream>>>(
      nq16, 0, E_, kw16, nullptr, 0, E_, k_b, k_proj, nullptr, EH_, 0, 0, E_, 0);
  k_gemm_wmma<<<dim3(EH_ / 16, ROWS_ / 16, 1), 32, 0, stream>>>(
      nq16, 0, E_, vw16, nullptr, 0, E_, v_b, v_proj, nullptr, EH_, 0, 0, E_, 0);

  // ---- 3) factorized score tables + V pack ----
  k_rowdots<<<cvblk(3 * BH_ * L_), 256, 0, stream>>>(
      q_proj, k_proj, e_proj, entity, attn_w, Aq, Bk, Ce);
  k_pack_ve<<<cvblk(BH_ * DH_ * L_), 256, 0, stream>>>(v_proj, e_proj, veT16);

  // ---- 4) scores + leaky_relu + softmax -> probs16 ----
  k_scores_softmax<<<dim3(BH_ * L_), 32, 0, stream>>>(Aq, Bk, Ce, attn_b, probs16);

  // ---- 5) attn = probs @ veT^T per head, scattered into a16 (L,B,E) ----
  k_gemm_wmma<<<dim3(DH_ / 16, L_ / 16, BH_), 32, 0, stream>>>(
      probs16, (long long)L_ * L_, L_, veT16, nullptr, (long long)DH_ * L_, L_,
      nullptr, nullptr, a16, E_ * B_, 0, /*cMode=*/1, L_, 0);

  // ---- 6) deconv as 2-tap GEMM: y = a@dW0^T + aprev@dW1^T + b -> y16 ----
  k_gemm_wmma<<<dim3(E_ / 16, ROWS_ / 16, 1), 32, 0, stream>>>(
      a16, 0, E_, dw0_16, dw1_16, 0, E_, deconv_b,
      nullptr, y16, E_, 0, 0, E_, B_);

  // ---- 7) out = y @ out_w^T + out_b -> d_out (fp32) ----
  k_gemm_wmma<<<dim3(E_ / 16, ROWS_ / 16, 1), 32, 0, stream>>>(
      y16, 0, E_, outw16, nullptr, 0, E_, out_b, out, nullptr, E_, 0, 0, E_, 0);
}
